// SVGGenerator_74199855006208
// MI455X (gfx1250) — matrix operand order repaired and checked by reading, not compiled, not run on hardware
//
#include <hip/hip_runtime.h>

// CDNA5 / gfx1250 vector types for WMMA
typedef _Float16 v16h __attribute__((ext_vector_type(16)));
typedef _Float16 v8h  __attribute__((ext_vector_type(8)));
typedef float    v8f  __attribute__((ext_vector_type(8)));

#define BB   4
#define RR   10
#define SP   2048      // S*P = 32*64
#define EPSF 1e-3f
#define HPAD 72        // padded LDS row stride (halfs): 144B, multiple of 16B
#define GPAD 66        // padded f32 row stride for gy

__global__ __launch_bounds__(128) void svg_warp_kernel(
    const float* __restrict__ points,    // [B,S,P,2] pixel coords
    const float* __restrict__ heatmap,   // [B,R,64,64]
    const float* __restrict__ s_shift,   // [B,R,2]
    const float* __restrict__ d_shift,   // [B,R,2]
    const float* __restrict__ s_affine,  // [B,R,2,2]
    const float* __restrict__ d_affine,  // [B,R,2,2]
    float* __restrict__ out)             // [B,S,P,2]
{
    __shared__ __align__(16) _Float16 sGx[64 * HPAD];  // gx f16, [spLocal][w]
    __shared__ __align__(16) _Float16 sHm[64 * HPAD];  // heatmap[b][r] f16, [h][w]
    __shared__ float sGy[64 * GPAD];                   // gy f32, [spLocal][h]
    __shared__ float sReg[RR][8];                      // m00,m01,m10,m11,ssx,ssy,dsx,dsy

    const int t      = threadIdx.x;
    const int b      = blockIdx.x >> 5;       // 32 sp-tiles per batch
    const int spBase = (blockIdx.x & 31) << 6; // 64 sp per block

    // ---- region params: affine_m = s_affine @ inv(d_affine), sign-fixed; pixel shifts
    if (t < RR) {
        const int   br = b * RR + t;
        const float* sa = s_affine + br * 4;
        const float* da = d_affine + br * 4;
        float a0 = da[0], a1 = da[1], a2 = da[2], a3 = da[3];
        float id = 1.0f / (a0 * a3 - a1 * a2);
        float i00 =  a3 * id, i01 = -a1 * id, i10 = -a2 * id, i11 = a0 * id;
        float s0 = sa[0], s1 = sa[1], s2 = sa[2], s3 = sa[3];
        float m00 = s0 * i00 + s1 * i10;
        float m01 = s0 * i01 + s1 * i11;
        float m10 = s2 * i00 + s3 * i10;
        float m11 = s2 * i01 + s3 * i11;
        float sg = (m00 >= 0.f) ? 1.f : -1.f;
        sReg[t][0] = m00 * sg; sReg[t][1] = m01 * sg;
        sReg[t][2] = m10 * sg; sReg[t][3] = m11 * sg;
        sReg[t][4] = (s_shift[br * 2 + 0] + 1.f) * 0.5f * 256.f;
        sReg[t][5] = (s_shift[br * 2 + 1] + 1.f) * 0.5f * 256.f;
        sReg[t][6] = (d_shift[br * 2 + 0] + 1.f) * 0.5f * 256.f;
        sReg[t][7] = (d_shift[br * 2 + 1] + 1.f) * 0.5f * 256.f;
    }

    // ---- separable Gaussian factors for this block's 64 points (heatmap coords = 0.25*pixel)
    for (int i = t; i < 64 * 64; i += 128) {
        int spL = i >> 6, w = i & 63;
        int sp  = spBase + spL;
        float px = points[(b * SP + sp) * 2 + 0] * 0.25f;
        float py = points[(b * SP + sp) * 2 + 1] * 0.25f;
        float dx = (float)w - px;
        float dy = (float)w - py;    // w doubles as h index for gy
        sGx[spL * HPAD + w] = (_Float16)__expf(-dx * dx);
        sGy[spL * GPAD + w] = __expf(-dy * dy);
    }
    __syncthreads();

    const int lane = t & 31;
    const int lo   = lane & 15;
    const int hi   = lane >> 4;
    const int wave = t >> 5;
    const int spL  = wave * 16 + lo;   // this lane's sp column within block
    const int sp   = spBase + spL;

    // ---- B fragments (gxᵀ, K=w, N=sp), ISA 16-bit B layout: lane=col, halfs 0..7 at K=8*hi, halfs 8..15 at K=16+8*hi
    v16h bg[2];
#pragma unroll
    for (int kk = 0; kk < 2; ++kk) {
        const _Float16* src = &sGx[spL * HPAD + kk * 32 + hi * 8];
        v8h x0 = *(const v8h*)(src);
        v8h x1 = *(const v8h*)(src + 16);
#pragma unroll
        for (int e = 0; e < 8; ++e) { bg[kk][e] = x0[e]; bg[kk][8 + e] = x1[e]; }
    }

    // ---- gy values this lane needs (C-layout rows: h = 16*ht + 8*hi + j), r-invariant
    float gyr[32];
#pragma unroll
    for (int ht = 0; ht < 4; ++ht)
#pragma unroll
        for (int j = 0; j < 8; ++j)
            gyr[ht * 8 + j] = sGy[spL * GPAD + ht * 16 + hi * 8 + j];

    const float ppx = points[(b * SP + sp) * 2 + 0];
    const float ppy = points[(b * SP + sp) * 2 + 1];
    float accX = 0.f, accY = 0.f, accD = 0.f;

    for (int r = 0; r < RR; ++r) {
        __syncthreads();  // previous iteration's A-fragment reads done
        const float* hsrc = heatmap + (size_t)(b * RR + r) * 4096;
        for (int i = t; i < 4096; i += 128)
            sHm[(i >> 6) * HPAD + (i & 63)] = (_Float16)hsrc[i];
        if (r + 1 < RR)
            __builtin_prefetch(hsrc + 4096 + t * 32, 0, 1);  // next region -> global_prefetch_b8
        __syncthreads();

        // T[h, sp] = heatmap[b,r] @ gxᵀ  via 4 h-tiles x 2 K-steps of v_wmma_f32_16x16x32_f16,
        // folded with gy on the fly: partial = sum_h T[h,sp] * gy[sp,h]
        float partial = 0.f;
#pragma unroll
        for (int ht = 0; ht < 4; ++ht) {
            v8f c = {};
#pragma unroll
            for (int kk = 0; kk < 2; ++kk) {
                // ISA 16-bit A layout: row M=lo; halfs 0..7 at K=8*hi, halfs 8..15 at K=16+8*hi
                const _Float16* arow = &sHm[(ht * 16 + lo) * HPAD + kk * 32 + hi * 8];
                v8h a0 = *(const v8h*)(arow);
                v8h a1 = *(const v8h*)(arow + 16);
                v16h af;
#pragma unroll
                for (int e = 0; e < 8; ++e) { af[e] = a0[e]; af[8 + e] = a1[e]; }
                c = __builtin_amdgcn_wmma_f32_16x16x32_f16(
                        /*neg_a=*/false, af, /*neg_b=*/false, bg[kk],
                        /*c_mod=*/(short)0, c, /*reuse_a=*/false, /*reuse_b=*/false);
            }
#pragma unroll
            for (int j = 0; j < 8; ++j) partial += c[j] * gyr[ht * 8 + j];
        }
        // lanes l and l+16 hold complementary row halves of the same sp column
        float infl = partial + __shfl_xor(partial, 16, 32);

        // warped point pm = (p - s_sh) @ M + d_sh ; accumulate weighted by influence
        float m00 = sReg[r][0], m01 = sReg[r][1], m10 = sReg[r][2], m11 = sReg[r][3];
        float ssx = sReg[r][4], ssy = sReg[r][5], dsx = sReg[r][6], dsy = sReg[r][7];
        float q0 = ppx - ssx, q1 = ppy - ssy;
        float pmx = q0 * m00 + q1 * m10 + dsx;
        float pmy = q0 * m01 + q1 * m11 + dsy;
        accX += infl * pmx;
        accY += infl * pmy;
        accD += infl;
    }

    if (hi == 0) {  // lanes 0-15 each own one sp
        float inv = 1.f / (accD + EPSF);
        out[(b * SP + sp) * 2 + 0] = (accX + EPSF * ppx) * inv;
        out[(b * SP + sp) * 2 + 1] = (accY + EPSF * ppy) * inv;
    }
}

extern "C" void kernel_launch(void* const* d_in, const int* in_sizes, int n_in,
                              void* d_out, int out_size, void* d_ws, size_t ws_size,
                              hipStream_t stream) {
    // setup_inputs() dict order: x, points, heatmap, s_shift, d_shift, s_affine, d_affine
    const float* points   = (const float*)d_in[1];
    const float* heatmap  = (const float*)d_in[2];
    const float* s_shift  = (const float*)d_in[3];
    const float* d_shift  = (const float*)d_in[4];
    const float* s_affine = (const float*)d_in[5];
    const float* d_affine = (const float*)d_in[6];
    float* out = (float*)d_out;
    (void)in_sizes; (void)n_in; (void)out_size; (void)d_ws; (void)ws_size;

    dim3 grid(BB * 32);   // 4 batches x 32 sp-tiles of 64
    dim3 block(128);      // 4 wave32s
    svg_warp_kernel<<<grid, block, 0, stream>>>(points, heatmap, s_shift, d_shift,
                                                s_affine, d_affine, out);
}
